// LSTMModel_75754633167016
// MI455X (gfx1250) — matrix operand order repaired and checked by reading, not compiled, not run on hardware
//
#include <hip/hip_runtime.h>
#include <math.h>

// ---------------- types ----------------
typedef __attribute__((ext_vector_type(16))) __bf16 v16bf;
typedef __attribute__((ext_vector_type(8)))  float  v8f;
typedef __attribute__((ext_vector_type(4)))  unsigned int v4u;

union Frag {
    v4u   q[2];   // 32 bytes
    v16bf v;      // 16 x bf16
};

__device__ __forceinline__ unsigned short f2bf(float f) {
    unsigned int u = __float_as_uint(f);
    unsigned int r = u + 0x7FFFu + ((u >> 16) & 1u);   // round-to-nearest-even
    return (unsigned short)(r >> 16);
}
__device__ __forceinline__ float bf2f(unsigned short h) {
    return __uint_as_float(((unsigned int)h) << 16);
}
__device__ __forceinline__ float sigm(float x) {
    return 1.0f / (1.0f + __expf(-x));
}

// ---------------- problem constants (match reference) ----------------
constexpr int Bc = 512;    // batch
constexpr int Tc = 512;    // time
constexpr int Hh = 30;     // hidden
// padded dims
constexpr int KP  = 64;    // padded input feature (E=50 or 2H=60 -> 64)
constexpr int NP  = 128;   // padded 4H gates (120 -> 128)
constexpr int HP  = 32;    // padded hidden (30 -> 32)

// =====================================================================
// Kernel 0: weight prep. Build bf16, zero-padded, frag-friendly images:
//   Wt  : [dir][n=128][k=64]   (N-major transposed input->gate weights)
//   Whht: [dir][n=128][k=32]   (N-major transposed recurrent weights)
//   bias: [dir][128] fp32
// Layer-1 input concat layout: fwd units at cols 0..29, bwd at 32..61.
// =====================================================================
__global__ void prep_weights(const float* __restrict__ Wih0, const float* __restrict__ Whh0,
                             const float* __restrict__ b0,
                             const float* __restrict__ Wih1, const float* __restrict__ Whh1,
                             const float* __restrict__ b1,
                             unsigned short* __restrict__ Wt0, unsigned short* __restrict__ Whht0,
                             unsigned short* __restrict__ Wt1, unsigned short* __restrict__ Whht1,
                             float* __restrict__ bias0, float* __restrict__ bias1)
{
    int idx = blockIdx.x * blockDim.x + threadIdx.x;

    if (idx < 2 * NP * KP) {                       // Wt0 / Wt1
        int k = idx & 63, n = (idx >> 6) & 127, d = idx >> 13;
        float v0 = (n < 120 && k < 50) ? Wih0[(d * 120 + n) * 50 + k] : 0.0f;
        Wt0[idx] = f2bf(v0);
        float v1 = 0.0f;
        if (n < 120) {
            int k2 = (k < 30) ? k : ((k >= 32 && k < 62) ? k - 2 : -1);
            if (k2 >= 0) v1 = Wih1[(d * 120 + n) * 60 + k2];
        }
        Wt1[idx] = f2bf(v1);
    }
    if (idx < 2 * NP * HP) {                       // Whht0 / Whht1
        int k = idx & 31, n = (idx >> 5) & 127, d = idx >> 12;
        float v0 = (n < 120 && k < Hh) ? Whh0[(d * 120 + n) * Hh + k] : 0.0f;
        float v1 = (n < 120 && k < Hh) ? Whh1[(d * 120 + n) * Hh + k] : 0.0f;
        Whht0[idx] = f2bf(v0);
        Whht1[idx] = f2bf(v1);
    }
    if (idx < 2 * NP) {                            // biases
        int n = idx & 127, d = idx >> 7;
        bias0[idx] = (n < 120) ? b0[d * 120 + n] : 0.0f;
        bias1[idx] = (n < 120) ? b1[d * 120 + n] : 0.0f;
    }
}

// =====================================================================
// Kernel 1: embedding gather -> bf16, row = t*B + b, 64 padded cols
// =====================================================================
__global__ void embed_bf16(const int* __restrict__ x, const float* __restrict__ emb,
                           unsigned short* __restrict__ Ebf)
{
    size_t idx = (size_t)blockIdx.x * blockDim.x + threadIdx.x;
    int    col = (int)(idx & 63);
    size_t row = idx >> 6;                    // t*B + b
    int b = (int)(row & (Bc - 1));
    int t = (int)(row >> 9);                  // Bc == 512
    float v = 0.0f;
    if (col < 50) {
        int tok = x[(size_t)b * Tc + t];
        v = emb[(size_t)tok * 50 + col];
    }
    Ebf[idx] = f2bf(v);
}

// =====================================================================
// Kernel 2: gx GEMM.  G[dir][row][128] = A[row][64] x W[64][128] + bias
// A bf16 row-major (rows = t*B + b), Wt bf16 N-major [dir][128][64].
// 128x128 block tile, 8 waves, wave w owns m-tile w, loops 8 n-tiles,
// K unrolled 2 x (16x16x32 bf16 WMMA). Result staged in LDS (bf16) and
// written out as coalesced b128 stores (G is the dominant HBM stream).
// =====================================================================
__global__ void gx_gemm(const unsigned short* __restrict__ A,
                        const unsigned short* __restrict__ Wt,
                        const float* __restrict__ bias,
                        unsigned short* __restrict__ G, int M)
{
    const int dir   = blockIdx.y;
    const int rbase = blockIdx.x * 128;
    const int tid   = threadIdx.x;

    __shared__ unsigned short sA[128][KP];    // 16 KB
    __shared__ unsigned short sW[NP][KP];     // 16 KB
    __shared__ unsigned short sC[128][NP];    // 32 KB  bf16 output staging

    // stage A tile (linear 16 KB) and W image (16 KB), 64 B per thread each
    {
        const v4u* gA = (const v4u*)(A + (size_t)rbase * KP);
        const v4u* gW = (const v4u*)(Wt + (size_t)dir * NP * KP);
        v4u* lA = (v4u*)&sA[0][0];
        v4u* lW = (v4u*)&sW[0][0];
#pragma unroll
        for (int i = 0; i < 4; ++i) {
            lA[tid + 256 * i] = gA[tid + 256 * i];
            lW[tid + 256 * i] = gW[tid + 256 * i];
        }
    }
    __syncthreads();

    const int wave  = tid >> 5;
    const int lane  = tid & 31;
    const int lhalf = lane >> 4;      // 0 | 1
    const int lmod  = lane & 15;
    const int m0    = wave * 16;

    for (int nt = 0; nt < 8; ++nt) {
        const int col0 = nt * 16;
        float bv = bias[dir * NP + col0 + lmod];
        v8f acc = { bv, bv, bv, bv, bv, bv, bv, bv };
#pragma unroll
        for (int ks = 0; ks < 2; ++ks) {
            Frag a, b;
            const unsigned short* ap = &sA[m0 + lmod][ks * 32 + lhalf * 8];
            a.q[0] = *(const v4u*)ap;
            a.q[1] = *(const v4u*)(ap + 16);
            const unsigned short* bp = &sW[col0 + lmod][ks * 32 + lhalf * 8];
            b.q[0] = *(const v4u*)bp;
            b.q[1] = *(const v4u*)(bp + 16);
            acc = __builtin_amdgcn_wmma_f32_16x16x32_bf16(
                false, a.v, false, b.v, (short)0, acc, false, false);
        }
        // scatter bf16 result to LDS staging (cheap ds writes)
#pragma unroll
        for (int v = 0; v < 8; ++v)
            sC[m0 + v + 8 * lhalf][col0 + lmod] = f2bf(acc[v]);
    }
    __syncthreads();

    // coalesced copy-out: 32 KB as b128 stores, 128 B per thread
    {
        unsigned short* gbase = G + ((size_t)dir * M + rbase) * NP;
        const v4u* src = (const v4u*)&sC[0][0];
        v4u*       dst = (v4u*)gbase;
#pragma unroll
        for (int i = 0; i < 8; ++i)
            dst[tid + 256 * i] = src[tid + 256 * i];
    }
}

// =====================================================================
// Kernel 3: LSTM scan. One block = 16 batch rows of one direction.
// Per step: h(bf16, LDS) x Whh(bf16, LDS) via 1 WMMA/wave (8 n-tiles),
// scatter fp32 gates to LDS, add precomputed gx (coalesced bf16, with
// global_prefetch of the next step's slice), fused activations + cell
// update (c in registers), h back to LDS.
// STORE_SEQ=1: write h sequence (layer 0). STORE_SEQ=0: running max.
// =====================================================================
template <int STORE_SEQ>
__global__ void lstm_scan(const unsigned short* __restrict__ G,     // [2][T*B][128]
                          const unsigned short* __restrict__ Whht,  // [2][128][32]
                          unsigned short* __restrict__ Hseq,        // [T*B][64] (layer0)
                          float* __restrict__ Hmax)                 // [B][64]   (layer1)
{
    const int dir = blockIdx.y;
    const int tid = threadIdx.x;

    __shared__ unsigned short hbuf[16][HP];   // 1 KB   h_{t-1} bf16
    __shared__ unsigned short sW[NP][HP];     // 8 KB   Whh^T
    __shared__ float gates[16][NP];           // 8 KB   fp32 gate pre-activations

    // stage Whh (N-major) and clear h
    {
        const v4u* gW = (const v4u*)(Whht + (size_t)dir * NP * HP);
        v4u* lW = (v4u*)&sW[0][0];
#pragma unroll
        for (int i = 0; i < 2; ++i) lW[tid + 256 * i] = gW[tid + 256 * i];
        unsigned short* hz = &hbuf[0][0];
        hz[tid] = 0; hz[tid + 256] = 0;
    }
    __syncthreads();

    const int wave  = tid >> 5;
    const int lane  = tid & 31;
    const int lhalf = lane >> 4;
    const int lmod  = lane & 15;
    const int col0  = wave * 16;              // wave -> its n-tile

    const int r    = tid >> 4;                // 0..15 : batch row within chunk
    const int brow = blockIdx.x * 16 + r;     // global batch row
    const int cb8  = (tid & 15) * 8;          // phase-B column chunk
    const int u0   = (tid & 15) * 2;          // phase-C unit pair

    const unsigned short* Gdir = G + (size_t)dir * Tc * Bc * NP;

    float creg[2] = { 0.0f, 0.0f };
    float hm[2]   = { -3.4e38f, -3.4e38f };

    for (int s = 0; s < Tc; ++s) {
        const int t = (dir == 0) ? s : (Tc - 1 - s);

        // ---- phase A: hWhh^T via WMMA (all lanes active, EXEC = ~0) ----
        Frag a, b;
        {
            const unsigned short* ap = &hbuf[lmod][lhalf * 8];
            a.q[0] = *(const v4u*)ap;
            a.q[1] = *(const v4u*)(ap + 16);
            const unsigned short* bp = &sW[col0 + lmod][lhalf * 8];
            b.q[0] = *(const v4u*)bp;
            b.q[1] = *(const v4u*)(bp + 16);
        }
        v8f acc = { 0.f, 0.f, 0.f, 0.f, 0.f, 0.f, 0.f, 0.f };
        acc = __builtin_amdgcn_wmma_f32_16x16x32_bf16(
            false, a.v, false, b.v, (short)0, acc, false, false);
#pragma unroll
        for (int v = 0; v < 8; ++v)
            gates[v + 8 * lhalf][col0 + lmod] = acc[v];
        __syncthreads();

        // ---- phase B: add gx (bias already folded in), coalesced bf16,
        //      and prefetch next step's slice (global_prefetch_b8) ----
        {
            const unsigned short* gsrc = Gdir + ((size_t)t * Bc + brow) * NP + cb8;
            union { v4u q; unsigned short h[8]; } gq;
            gq.q = *(const v4u*)gsrc;
            if (s + 1 < Tc) {
                const int tn = (dir == 0) ? t + 1 : t - 1;
                __builtin_prefetch(Gdir + ((size_t)tn * Bc + brow) * NP + cb8, 0, 1);
            }
#pragma unroll
            for (int j = 0; j < 8; ++j)
                gates[r][cb8 + j] += bf2f(gq.h[j]);
        }
        __syncthreads();

        // ---- phase C: activations + cell update, h -> LDS (+ out) ----
#pragma unroll
        for (int k = 0; k < 2; ++k) {
            const int u = u0 + k;
            float hval = 0.0f;
            if (u < Hh) {
                float gi = gates[r][u];
                float gf = gates[r][Hh + u];
                float gg = gates[r][2 * Hh + u];
                float go = gates[r][3 * Hh + u];
                float c  = sigm(gf) * creg[k] + sigm(gi) * tanhf(gg);
                creg[k]  = c;
                hval     = sigm(go) * tanhf(c);
            }
            hbuf[r][u] = f2bf(hval);
            if (STORE_SEQ) {
                Hseq[((size_t)t * Bc + brow) * KP + dir * 32 + u] = f2bf(hval);
            } else {
                hm[k] = fmaxf(hm[k], hval);
            }
        }
        __syncthreads();
    }

    if (!STORE_SEQ) {
#pragma unroll
        for (int k = 0; k < 2; ++k) {
            const int u = u0 + k;
            if (u < Hh) Hmax[(size_t)brow * KP + dir * Hh + u] = hm[k];
        }
    }
}

// =====================================================================
// Kernel 4: head MLP. out = sigmoid(fc2(relu(fc1(hmax))))  (3 MFLOP)
// =====================================================================
__global__ void head_mlp(const float* __restrict__ Hmax,
                         const float* __restrict__ fc1w, const float* __restrict__ fc1b,
                         const float* __restrict__ fc2w, const float* __restrict__ fc2b,
                         float* __restrict__ out)
{
    int b = blockIdx.x * blockDim.x + threadIdx.x;
    if (b >= Bc) return;
    const float* hv = Hmax + (size_t)b * KP;
    float hidden[50];
#pragma unroll 1
    for (int j = 0; j < 50; ++j) {
        float s = fc1b[j];
        const float* w = fc1w + j * 60;
        for (int k = 0; k < 60; ++k) s += w[k] * hv[k];
        hidden[j] = fmaxf(s, 0.0f);
    }
#pragma unroll 1
    for (int j = 0; j < 6; ++j) {
        float s = fc2b[j];
        const float* w = fc2w + j * 50;
        for (int k = 0; k < 50; ++k) s += w[k] * hidden[k];
        out[b * 6 + j] = 1.0f / (1.0f + __expf(-s));
    }
}

// =====================================================================
// launcher
// =====================================================================
extern "C" void kernel_launch(void* const* d_in, const int* in_sizes, int n_in,
                              void* d_out, int out_size, void* d_ws, size_t ws_size,
                              hipStream_t stream)
{
    (void)in_sizes; (void)n_in; (void)out_size; (void)ws_size;

    const int*   x    = (const int*)  d_in[0];
    const float* emb  = (const float*)d_in[1];
    const float* Wih0 = (const float*)d_in[2];
    const float* Whh0 = (const float*)d_in[3];
    const float* b0   = (const float*)d_in[4];
    const float* Wih1 = (const float*)d_in[5];
    const float* Whh1 = (const float*)d_in[6];
    const float* b1   = (const float*)d_in[7];
    const float* fc1w = (const float*)d_in[8];
    const float* fc1b = (const float*)d_in[9];
    const float* fc2w = (const float*)d_in[10];
    const float* fc2b = (const float*)d_in[11];

    const int M = Tc * Bc;   // 262144 rows

    char*  ws  = (char*)d_ws;
    size_t off = 0;
    auto alloc = [&](size_t bytes) -> char* {
        char* p = ws + off;
        off += (bytes + 255) & ~(size_t)255;
        return p;
    };
    unsigned short* Ebf   = (unsigned short*)alloc((size_t)M * KP * 2);        // 32 MiB
    unsigned short* G     = (unsigned short*)alloc((size_t)2 * M * NP * 2);    // 128 MiB
    unsigned short* H0    = (unsigned short*)alloc((size_t)M * KP * 2);        // 32 MiB
    float*          Hmax  = (float*)         alloc((size_t)Bc * KP * 4);
    unsigned short* Wt0   = (unsigned short*)alloc((size_t)2 * NP * KP * 2);
    unsigned short* Whht0 = (unsigned short*)alloc((size_t)2 * NP * HP * 2);
    unsigned short* Wt1   = (unsigned short*)alloc((size_t)2 * NP * KP * 2);
    unsigned short* Whht1 = (unsigned short*)alloc((size_t)2 * NP * HP * 2);
    float*          bias0 = (float*)         alloc((size_t)2 * NP * 4);
    float*          bias1 = (float*)         alloc((size_t)2 * NP * 4);

    // 0) weight prep (bf16, padded, N-major transposed)
    prep_weights<<<64, 256, 0, stream>>>(Wih0, Whh0, b0, Wih1, Whh1, b1,
                                         Wt0, Whht0, Wt1, Whht1, bias0, bias1);
    // 1) embedding gather -> bf16 padded rows (t*B + b)
    embed_bf16<<<(int)(((size_t)M * KP) / 256), 256, 0, stream>>>(x, emb, Ebf);
    // 2) layer-0 gx GEMM (both directions via grid.y)
    gx_gemm<<<dim3(M / 128, 2), 256, 0, stream>>>(Ebf, Wt0, bias0, G, M);
    // 3) layer-0 scan, store hidden sequence
    lstm_scan<1><<<dim3(Bc / 16, 2), 256, 0, stream>>>(G, Whht0, H0, nullptr);
    // 4) layer-1 gx GEMM (reuses G)
    gx_gemm<<<dim3(M / 128, 2), 256, 0, stream>>>(H0, Wt1, bias1, G, M);
    // 5) layer-1 scan with fused max-over-time
    lstm_scan<0><<<dim3(Bc / 16, 2), 256, 0, stream>>>(G, Whht1, nullptr, Hmax);
    // 6) head MLP -> sigmoid output [B, 6]
    head_mlp<<<2, 256, 0, stream>>>(Hmax, fc1w, fc1b, fc2w, fc2b, (float*)d_out);
}
